// KineticForecastingFramework_78881369359043
// MI455X (gfx1250) — matrix operand order repaired and checked by reading, not compiled, not run on hardware
//
#include <hip/hip_runtime.h>
#include <math.h>

#define QDIM 64
#define HDIM 64
#define DT_C 0.1f
#define REG_C 1e-6f

typedef __attribute__((ext_vector_type(16))) _Float16 v16h;
typedef __attribute__((ext_vector_type(8)))  float    v8f;

#define XS 72   // activation staging row stride (halves): conflict-free, 16B aligned runs
#define WS 68   // transposed-weight row stride (halves): conflict-free

__device__ __forceinline__ void atomAddF(float* p, float v) {
    (void)__hip_atomic_fetch_add(p, v, __ATOMIC_RELAXED, __HIP_MEMORY_SCOPE_AGENT);
}

// ---------------------------------------------------------------- zero scratch
__global__ void kff_zero(float* __restrict__ trans, float* __restrict__ outdeg, int n) {
    int i = blockIdx.x * blockDim.x + threadIdx.x;
    int nq = n * QDIM;
    if (i < nq)            trans[i] = 0.f;
    else if (i < nq + n)   outdeg[i - nq] = 0.f;
}

// ---------------------------------------------------------------- out-degree
__global__ void kff_deg(const int* __restrict__ esrc, float* __restrict__ outdeg, int E) {
    int e = blockIdx.x * blockDim.x + threadIdx.x;
    if (e < E) atomAddF(&outdeg[esrc[e]], 1.0f);
}

// ---------------------------------------------------------------- transport (graph scatter)
// 16 threads per edge, float4 per thread. trans[src] += m, trans[dst] -= m
__global__ void kff_edges(const float* __restrict__ f,
                          const int* __restrict__ esrc, const int* __restrict__ edst,
                          const float* __restrict__ ew, const float* __restrict__ xi,
                          const float* __restrict__ outdeg, float* __restrict__ trans, int E) {
    int idx = blockIdx.x * blockDim.x + threadIdx.x;
    int e = idx >> 4;
    if (e >= E) return;
    int j = idx & 15;
    int s = esrc[e], d = edst[e];
    float qe = ew[e] / outdeg[s];
    const float4* f4 = (const float4*)f;
    float4 fs = f4[(size_t)s * 16 + j];
    float4 fd = f4[(size_t)d * 16 + j];
    float4 x4 = ((const float4*)xi)[j];
    float mx = qe * x4.x * (fmaxf(fd.x, 0.f) - fmaxf(fs.x, 0.f));
    float my = qe * x4.y * (fmaxf(fd.y, 0.f) - fmaxf(fs.y, 0.f));
    float mz = qe * x4.z * (fmaxf(fd.z, 0.f) - fmaxf(fs.z, 0.f));
    float mw = qe * x4.w * (fmaxf(fd.w, 0.f) - fmaxf(fs.w, 0.f));
    float* ts = trans + (size_t)s * QDIM + j * 4;
    float* td = trans + (size_t)d * QDIM + j * 4;
    atomAddF(ts + 0,  mx); atomAddF(ts + 1,  my); atomAddF(ts + 2,  mz); atomAddF(ts + 3,  mw);
    atomAddF(td + 0, -mx); atomAddF(td + 1, -my); atomAddF(td + 2, -mz); atomAddF(td + 3, -mw);
}

// ---------------------------------------------------------------- fused MLP + projection
// 256 threads = 8 waves; each wave owns a 16-row tile (128 rows / WG).
__global__ __launch_bounds__(256) void kff_mlp(
        const float* __restrict__ f, const float* __restrict__ macro_u,
        const float* __restrict__ xi,
        const float* __restrict__ w_in, const float* __restrict__ b_in,
        const float* __restrict__ w_hid, const float* __restrict__ b_hid,
        const float* __restrict__ w_out, const float* __restrict__ b_out,
        float* __restrict__ omega, int Nn) {
    __shared__ _Float16 sWt[5 * 64 * WS];                  // weights transposed [l][n][k]
    __shared__ float    sB[5 * 64];
    __shared__ float    sXi[64];
    __shared__ __align__(16) _Float16 sX[8 * 16 * XS];     // per-wave activation staging

    const int tid  = threadIdx.x;
    const int wave = tid >> 5;
    const int lane = tid & 31;
    const int n    = lane & 15;
    const int hi   = lane >> 4;

    // stage weights (f32 -> f16, transposed) and biases
    for (int i = tid; i < 5 * 4096; i += 256) {
        int l = i >> 12, r = i & 4095;
        int k = r >> 6,  nn = r & 63;
        float v;
        if (l == 0)      v = w_in[r];
        else if (l <= 3) v = w_hid[(l - 1) * 4096 + r];
        else             v = w_out[r];
        sWt[l * 64 * WS + nn * WS + k] = (_Float16)v;
    }
    for (int i = tid; i < 5 * 64; i += 256) {
        int l = i >> 6, nn = i & 63;
        float v;
        if (l == 0)      v = b_in[nn];
        else if (l <= 3) v = b_hid[(l - 1) * 64 + nn];
        else             v = b_out[nn];
        sB[i] = v;
    }
    if (tid < 64) sXi[tid] = xi[tid];
    __syncthreads();

    const int r0 = (blockIdx.x * 8 + wave) * 16;
    _Float16* xb = sX + wave * 16 * XS;

    // stage layer-0 input: clamp(f,0) -> f16 staging buffer
    {
        int grow = r0 + n; if (grow > Nn - 1) grow = Nn - 1;
        const float* src = f + (size_t)grow * QDIM + hi * 32;
        _Float16* dstp = xb + n * XS + hi * 32;
        #pragma unroll
        for (int c = 0; c < 32; c += 4) {
            float4 v = *(const float4*)(src + c);
            dstp[c + 0] = (_Float16)fmaxf(v.x, 0.f);
            dstp[c + 1] = (_Float16)fmaxf(v.y, 0.f);
            dstp[c + 2] = (_Float16)fmaxf(v.z, 0.f);
            dstp[c + 3] = (_Float16)fmaxf(v.w, 0.f);
        }
    }
    __syncthreads();

    v8f acc[4];
    for (int l = 0; l < 5; ++l) {
        // A fragments: lane holds row n, K runs at 8*hi / 16+8*hi (ISA 16-bit A layout)
        v16h a0, a1;
        {
            const _Float16* row = xb + n * XS + hi * 8;
            #pragma unroll
            for (int h = 0; h < 8; ++h) { a0[h] = row[h];      a0[8 + h] = row[16 + h]; }
            #pragma unroll
            for (int h = 0; h < 8; ++h) { a1[h] = row[32 + h]; a1[8 + h] = row[48 + h]; }
        }
        const _Float16* wl = sWt + l * 64 * WS;
        #pragma unroll
        for (int t = 0; t < 4; ++t) {
            float bn = sB[l * 64 + t * 16 + n];
            v8f c;
            #pragma unroll
            for (int j = 0; j < 8; ++j) c[j] = bn;
            // B fragments: lane holds col n, K runs at 8*hi / 16+8*hi (weights stored [n][k])
            const _Float16* col = wl + (t * 16 + n) * WS + hi * 8;
            v16h b0, b1;
            #pragma unroll
            for (int h = 0; h < 16; ++h) { int kk = (h < 8) ? h : h + 8; b0[h] = col[kk]; b1[h] = col[32 + kk]; }
            c = __builtin_amdgcn_wmma_f32_16x16x32_f16(false, a0, false, b0, (short)0, c, false, false);
            c = __builtin_amdgcn_wmma_f32_16x16x32_f16(false, a1, false, b1, (short)0, c, false, false);
            acc[t] = c;
        }
        if (l < 4) {
            __syncthreads();
            #pragma unroll
            for (int t = 0; t < 4; ++t) {
                #pragma unroll
                for (int j = 0; j < 8; ++j) {
                    int m = j + 8 * hi;                       // D layout: M = j + 8*hi, N = lane%16
                    xb[m * XS + t * 16 + n] = (_Float16)fmaxf(acc[t][j], 0.f);
                }
            }
            __syncthreads();
        }
    }

    // tanh output
    #pragma unroll
    for (int t = 0; t < 4; ++t) {
        #pragma unroll
        for (int j = 0; j < 8; ++j) acc[t][j] = tanhf(acc[t][j]);
    }

    // conservation projection: CCT closed-form from moments of xi; v0,v1 via lane reduction
    float Sxi = 0.f, Sxi2 = 0.f;
    #pragma unroll
    for (int q = 0; q < 64; ++q) { float x = sXi[q]; Sxi += x; Sxi2 += x * x; }

    float u[8];
    #pragma unroll
    for (int j = 0; j < 8; ++j) {
        int row = r0 + j + 8 * hi; if (row > Nn - 1) row = Nn - 1;
        u[j] = macro_u[row];
    }

    float v0[8], v1[8];
    #pragma unroll
    for (int j = 0; j < 8; ++j) { v0[j] = 0.f; v1[j] = 0.f; }
    #pragma unroll
    for (int t = 0; t < 4; ++t) {
        #pragma unroll
        for (int j = 0; j < 8; ++j) {
            float om = acc[t][j];
            float xs = sXi[t * 16 + n] + u[j];
            v0[j] += om;
            v1[j] += xs * om;
        }
    }
    #pragma unroll
    for (int j = 0; j < 8; ++j) {
        #pragma unroll
        for (int off = 1; off < 16; off <<= 1) {
            v0[j] += __shfl_xor(v0[j], off, 16);
            v1[j] += __shfl_xor(v1[j], off, 16);
        }
    }

    const float w = 1.0f / 64.0f, w2 = w * w;
    float s0[8], s1[8];
    #pragma unroll
    for (int j = 0; j < 8; ++j) {
        float a01 = w2 * Sxi + w * u[j];
        float a11 = w2 * Sxi2 + 2.f * u[j] * w2 * Sxi + w * u[j] * u[j] + REG_C;
        float a00 = w + REG_C;
        float vv0 = w * v0[j], vv1 = w * v1[j];
        float inv = 1.f / (a00 * a11 - a01 * a01);
        s0[j] = (a11 * vv0 - a01 * vv1) * inv;
        s1[j] = (a00 * vv1 - a01 * vv0) * inv;
    }
    #pragma unroll
    for (int t = 0; t < 4; ++t) {
        #pragma unroll
        for (int j = 0; j < 8; ++j) {
            int row = r0 + j + 8 * hi;
            if (row < Nn) {
                int q = t * 16 + n;
                float xs = sXi[q] + u[j];
                omega[(size_t)row * QDIM + q] = acc[t][j] - w * (s0[j] + xs * s1[j]);
            }
        }
    }
}

// ---------------------------------------------------------------- final update
__global__ void kff_final(const float* __restrict__ f, const float* __restrict__ trans,
                          const float* __restrict__ omega, const float* __restrict__ source,
                          float* __restrict__ out, int n) {
    int i = blockIdx.x * blockDim.x + threadIdx.x;   // float4 index over n*16
    if (i >= n * 16) return;
    float4 a = ((const float4*)f)[i];
    float4 t = ((const float4*)trans)[i];
    float4 o = ((const float4*)omega)[i];
    float4 s = ((const float4*)source)[i];
    float4 r;
    r.x = fmaxf(fmaxf(a.x, 0.f) - DT_C * (t.x - o.x - s.x), 0.f);
    r.y = fmaxf(fmaxf(a.y, 0.f) - DT_C * (t.y - o.y - s.y), 0.f);
    r.z = fmaxf(fmaxf(a.z, 0.f) - DT_C * (t.z - o.z - s.z), 0.f);
    r.w = fmaxf(fmaxf(a.w, 0.f) - DT_C * (t.w - o.w - s.w), 0.f);
    ((float4*)out)[i] = r;
}

// ---------------------------------------------------------------- launch
extern "C" void kernel_launch(void* const* d_in, const int* in_sizes, int n_in,
                              void* d_out, int out_size, void* d_ws, size_t ws_size,
                              hipStream_t stream) {
    (void)n_in; (void)out_size; (void)ws_size;
    const float* f       = (const float*)d_in[0];
    const float* macro_u = (const float*)d_in[1];
    const float* source  = (const float*)d_in[2];
    const int*   esrc    = (const int*)d_in[3];
    const int*   edst    = (const int*)d_in[4];
    const float* ew      = (const float*)d_in[5];
    const float* xi      = (const float*)d_in[6];
    const float* w_in    = (const float*)d_in[7];
    const float* b_in    = (const float*)d_in[8];
    const float* w_hid   = (const float*)d_in[9];
    const float* b_hid   = (const float*)d_in[10];
    const float* w_out   = (const float*)d_in[11];
    const float* b_out   = (const float*)d_in[12];

    const int Nn = in_sizes[1];   // macro_u is [N,1]
    const int E  = in_sizes[3];

    float* out   = (float*)d_out;
    size_t nq    = (size_t)Nn * QDIM;
    float* omega  = (float*)d_ws;
    float* trans  = omega + nq;
    float* outdeg = trans + nq;

    int zc = Nn * (QDIM + 1);
    kff_zero<<<(zc + 255) / 256, 256, 0, stream>>>(trans, outdeg, Nn);
    kff_deg<<<(E + 255) / 256, 256, 0, stream>>>(esrc, outdeg, E);
    kff_mlp<<<(Nn + 127) / 128, 256, 0, stream>>>(f, macro_u, xi, w_in, b_in,
                                                  w_hid, b_hid, w_out, b_out,
                                                  omega, Nn);
    long long et = (long long)E * 16;
    kff_edges<<<(int)((et + 255) / 256), 256, 0, stream>>>(f, esrc, edst, ew, xi,
                                                           outdeg, trans, E);
    long long ft = (long long)Nn * 16;
    kff_final<<<(int)((ft + 255) / 256), 256, 0, stream>>>(f, trans, omega, source, out, Nn);
}
